// MultiHeadAttention_51488067945128
// MI455X (gfx1250) — matrix operand order repaired and checked by reading, not compiled, not run on hardware
//
#include <hip/hip_runtime.h>

// MHA forward for MI455X (gfx1250), wave32 + v_wmma_f32_16x16x32_f16,
// double-buffered LDS pipelines + async global->LDS staging (ASYNCcnt).
// B=2, S=2048, D=1024, H=16, dh=64 (hardcoded from the reference).

typedef __attribute__((ext_vector_type(16))) _Float16 v16h;
typedef __attribute__((ext_vector_type(8)))  _Float16 v8h;
typedef __attribute__((ext_vector_type(8)))  float    v8f;

#define D_MODEL 1024
#define SEQ     2048
#define NH      16
#define DH      64
#define BATCH   2
#define MTOT    (BATCH*SEQ)   // 4096

// ---- CDNA5 async global->LDS (tracked by ASYNCcnt) ------------------------
// global_load_async_to_lds_b128: VDST = LDS byte offset (VGPR), VADDR = 64-bit
// global address. Flat->LDS aperture keeps the LDS offset in addr[31:0], so
// truncating the generic pointer of a __shared__ object gives the DS offset.
__device__ __forceinline__ void async_load_lds_b128(const void* gptr, void* lptr) {
    const unsigned ldsOff = (unsigned)(uintptr_t)lptr;
    asm volatile("global_load_async_to_lds_b128 %0, %1, off"
                 :: "v"(ldsOff), "v"(gptr) : "memory");
}
__device__ __forceinline__ void wait_asynccnt0() {
    asm volatile("s_wait_asynccnt 0x0" ::: "memory");
}

// ---------------------------------------------------------------------------
// GEMM: C[M,N] = A[M,K] @ W[N,K]^T + bias[N]
//   A, W fp32 in global; fp32->fp16 conversion happens in the staging pipe.
//   MODE 0: output fp16, head-split layout [B,H,S,64] (projections)
//   MODE 1: output fp32, plain [M,N]                  (final projection)
// 256 threads (8 waves), block tile 128(M) x 64(N), K-step 32, double-buffered
// LDS with register staging: one barrier per K-step, loads issued before the
// WMMA block so their latency hides under the matrix math.
// ---------------------------------------------------------------------------
template<int MODE>
__global__ __launch_bounds__(256)
void gemm_wmma_kernel(const float* __restrict__ A, const float* __restrict__ W,
                      const float* __restrict__ bias, void* __restrict__ outp)
{
    __shared__ __align__(32) _Float16 As[2][128 * 32];
    __shared__ __align__(32) _Float16 Ws[2][64 * 32];

    const int tid  = threadIdx.x;
    const int lane = tid & 31;
    const int wv   = tid >> 5;       // wave 0..7
    const int r    = lane & 15;
    const int hf   = lane >> 4;      // half of wave (0/1)

    const int row0 = blockIdx.y * 128;   // M tile
    const int n0b  = blockIdx.x * 64;    // N tile

    const int arow = tid >> 1;            // 0..127
    const int acol = (tid & 1) * 16;
    const int wrow = (tid & 127) >> 1;    // 0..63
    const int wcol = (tid & 1) * 16;
    const bool doW = tid < 128;

    const float* aptr = A + (size_t)(row0 + arow) * D_MODEL + acol;
    const float* wptr = W + (size_t)(n0b + wrow) * D_MODEL + wcol;

    float areg[16], wreg[16];

    auto stageLoad = [&](int k0) {
        #pragma unroll
        for (int i = 0; i < 16; ++i) areg[i] = aptr[k0 + i];
        if (doW) {
            #pragma unroll
            for (int i = 0; i < 16; ++i) wreg[i] = wptr[k0 + i];
        }
    };
    auto stageStore = [&](int buf) {
        _Float16* ad = &As[buf][arow * 32 + acol];
        #pragma unroll
        for (int i = 0; i < 16; ++i) ad[i] = (_Float16)areg[i];
        if (doW) {
            _Float16* wd = &Ws[buf][wrow * 32 + wcol];
            #pragma unroll
            for (int i = 0; i < 16; ++i) wd[i] = (_Float16)wreg[i];
        }
    };

    stageLoad(0);
    stageStore(0);
    __syncthreads();

    v8f acc[4] = {};

    for (int it = 0; it < D_MODEL / 32; ++it) {
        const int  cur  = it & 1;
        const bool more = (it + 1) < D_MODEL / 32;
        if (more) stageLoad((it + 1) * 32);   // issue next-slab global loads early

        // A fragment: 16x32, rows = wv*16 + (lane&15); elements K = 16*hf + e
        // (contiguous, matching B).
        v16h af;
        {
            const v8h g0 = *(const v8h*)&As[cur][(wv * 16 + r) * 32 + 16 * hf];
            const v8h g1 = *(const v8h*)&As[cur][(wv * 16 + r) * 32 + 16 * hf + 8];
            #pragma unroll
            for (int i = 0; i < 8; ++i) { af[i] = g0[i]; af[i + 8] = g1[i]; }
        }
        // B fragments (32x16): lane col = n, elements K = 16*hf + e (contiguous)
        #pragma unroll
        for (int t = 0; t < 4; ++t) {
            const v16h bf = *(const v16h*)&Ws[cur][(t * 16 + r) * 32 + 16 * hf];
            acc[t] = __builtin_amdgcn_wmma_f32_16x16x32_f16(
                false, af, false, bf, (short)0, acc[t], false, false);
        }

        if (more) stageStore(cur ^ 1);        // commit next slab after the math
        __syncthreads();
    }

    // store (C layout: element rr -> row M = 8*hf + rr, col N = lane&15)
    #pragma unroll
    for (int t = 0; t < 4; ++t) {
        #pragma unroll
        for (int rr = 0; rr < 8; ++rr) {
            const int m = row0 + wv * 16 + hf * 8 + rr;
            const int n = n0b + t * 16 + r;
            const float val = acc[t][rr] + bias[n];
            if (MODE == 0) {
                _Float16* out = (_Float16*)outp;
                const int b = m >> 11, s = m & 2047;     // m = b*SEQ + s
                const int h = n >> 6,  dd = n & 63;      // n = h*64 + dd
                out[(((size_t)(b * NH + h)) * SEQ + s) * DH + dd] = (_Float16)val;
            } else {
                ((float*)outp)[(size_t)m * D_MODEL + n] = val;
            }
        }
    }
}

// ---------------------------------------------------------------------------
// Flash attention: one block per (b,h, 64-query tile). 128 threads = 4 waves,
// each wave owns 16 query rows; streams 64-key chunks with online softmax.
//   scores = (Q @ K^T) * 0.125 ; P = softmax ; O = P @ V
// K tiles are staged with global_load_async_to_lds_b128 (ASYNCcnt) into a
// double buffer; V is read coalesced into registers and scattered transposed
// into the alternate LDS buffer after the WMMA block. One barrier per chunk.
// ---------------------------------------------------------------------------
__global__ __launch_bounds__(128)
void attn_wmma_kernel(const _Float16* __restrict__ Qp, const _Float16* __restrict__ Kp,
                      const _Float16* __restrict__ Vp, float* __restrict__ Op)
{
    __shared__ __align__(32) _Float16 Kt[2][64 * 64];   // [key][d]
    __shared__ __align__(32) _Float16 VtT[2][64 * 64];  // transposed: [d][key]
    __shared__ __align__(32) _Float16 Pt[4 * 16 * 64];  // per-wave P tiles

    const int tid  = threadIdx.x;
    const int lane = tid & 31;
    const int wv   = tid >> 5;     // 0..3
    const int r    = lane & 15;
    const int hf   = lane >> 4;

    const int qt = blockIdx.x & 31;        // S/64 = 32 query tiles
    const int bh = blockIdx.x >> 5;        // b*NH + h, 0..31
    const size_t headOff = (size_t)bh * SEQ * DH;
    const int qbase = qt * 64;

    // ---- preload Q A-fragments (rows qbase + wv*16 + r, d = 0..63) ----
    const _Float16* qrow = Qp + headOff + (size_t)(qbase + wv * 16 + r) * DH;
    v16h aq[2];
    #pragma unroll
    for (int c = 0; c < 2; ++c) {
        const v8h g0 = *(const v8h*)(qrow + c * 32 + 16 * hf);
        const v8h g1 = *(const v8h*)(qrow + c * 32 + 16 * hf + 8);
        #pragma unroll
        for (int i = 0; i < 8; ++i) { aq[c][i] = g0[i]; aq[c][i + 8] = g1[i]; }
    }

    // ---- staging helpers: each thread owns K/V row krow, 32-col half cc ----
    const int krow = tid >> 1;
    const int cc   = (tid & 1) * 32;
    const _Float16* kbaseP = Kp + headOff + (size_t)krow * DH + cc;
    const _Float16* vbaseP = Vp + headOff + (size_t)krow * DH + cc;

    auto issueK = [&](int buf, int kb) {    // async copy, no VGPR round-trip
        #pragma unroll
        for (int c = 0; c < 4; ++c)
            async_load_lds_b128(kbaseP + (size_t)kb * DH + 8 * c,
                                &Kt[buf][krow * 64 + cc + 8 * c]);
    };
    v8h vreg[4];
    auto loadV = [&](int kb) {
        #pragma unroll
        for (int c = 0; c < 4; ++c)
            vreg[c] = *(const v8h*)(vbaseP + (size_t)kb * DH + 8 * c);
    };
    auto scatterV = [&](int buf) {          // transpose into [d][key]
        #pragma unroll
        for (int c = 0; c < 4; ++c)
            #pragma unroll
            for (int i = 0; i < 8; ++i)
                VtT[buf][(cc + 8 * c + i) * 64 + krow] = vreg[c][i];
    };

    float mrow[8], lrow[8];
    v8f oacc[4] = {};
    #pragma unroll
    for (int i = 0; i < 8; ++i) { mrow[i] = -3.0e38f; lrow[i] = 0.0f; }

    _Float16* Pw = &Pt[wv * 16 * 64];

    // ---- prologue: stage chunk 0 ----
    issueK(0, 0);
    loadV(0);
    scatterV(0);
    wait_asynccnt0();
    __syncthreads();

    for (int it = 0; it < SEQ / 64; ++it) {
        const int  cur  = it & 1;
        const bool more = (it + 1) < SEQ / 64;
        if (more) {                       // issue next chunk before the math
            issueK(cur ^ 1, (it + 1) * 64);
            loadV((it + 1) * 64);
        }

        // ---- scores: 16 q-rows x 64 keys (4 tiles, K-dim = d = 64) ----
        v8f sc[4] = {};
        #pragma unroll
        for (int t = 0; t < 4; ++t) {
            const v16h b0 = *(const v16h*)&Kt[cur][(t * 16 + r) * 64 + 16 * hf];
            const v16h b1 = *(const v16h*)&Kt[cur][(t * 16 + r) * 64 + 32 + 16 * hf];
            sc[t] = __builtin_amdgcn_wmma_f32_16x16x32_f16(false, aq[0], false, b0, (short)0, sc[t], false, false);
            sc[t] = __builtin_amdgcn_wmma_f32_16x16x32_f16(false, aq[1], false, b1, (short)0, sc[t], false, false);
        }
        #pragma unroll
        for (int t = 0; t < 4; ++t)
            #pragma unroll
            for (int i = 0; i < 8; ++i) sc[t][i] *= 0.125f;   // 1/sqrt(64)

        // ---- online softmax (row = 8*hf + i; reduce 16 lanes of the half) ----
        float mnew[8], alpha[8];
        #pragma unroll
        for (int i = 0; i < 8; ++i) {
            float v = fmaxf(fmaxf(sc[0][i], sc[1][i]), fmaxf(sc[2][i], sc[3][i]));
            v = fmaxf(v, __shfl_xor(v, 1, 32));
            v = fmaxf(v, __shfl_xor(v, 2, 32));
            v = fmaxf(v, __shfl_xor(v, 4, 32));
            v = fmaxf(v, __shfl_xor(v, 8, 32));
            mnew[i]  = fmaxf(mrow[i], v);
            alpha[i] = __expf(mrow[i] - mnew[i]);
        }
        float rs[8] = {};
        #pragma unroll
        for (int t = 0; t < 4; ++t) {
            #pragma unroll
            for (int i = 0; i < 8; ++i) {
                const float p = __expf(sc[t][i] - mnew[i]);
                rs[i] += p;
                Pw[(hf * 8 + i) * 64 + t * 16 + r] = (_Float16)p;  // C->A via LDS
            }
        }
        #pragma unroll
        for (int i = 0; i < 8; ++i) {
            rs[i] += __shfl_xor(rs[i], 1, 32);
            rs[i] += __shfl_xor(rs[i], 2, 32);
            rs[i] += __shfl_xor(rs[i], 4, 32);
            rs[i] += __shfl_xor(rs[i], 8, 32);
            lrow[i] = lrow[i] * alpha[i] + rs[i];
            mrow[i] = mnew[i];
        }
        #pragma unroll
        for (int t = 0; t < 4; ++t)
            #pragma unroll
            for (int i = 0; i < 8; ++i) oacc[t][i] *= alpha[i];

        // ---- O += P @ V (P A-frags from per-wave LDS; same-wave DS is in-order)
        v16h ap[2];
        #pragma unroll
        for (int c = 0; c < 2; ++c) {
            const v8h g0 = *(const v8h*)&Pw[r * 64 + c * 32 + 16 * hf];
            const v8h g1 = *(const v8h*)&Pw[r * 64 + c * 32 + 16 * hf + 8];
            #pragma unroll
            for (int i = 0; i < 8; ++i) { ap[c][i] = g0[i]; ap[c][i + 8] = g1[i]; }
        }
        #pragma unroll
        for (int t = 0; t < 4; ++t) {
            const v16h b0 = *(const v16h*)&VtT[cur][(t * 16 + r) * 64 + 16 * hf];
            const v16h b1 = *(const v16h*)&VtT[cur][(t * 16 + r) * 64 + 32 + 16 * hf];
            oacc[t] = __builtin_amdgcn_wmma_f32_16x16x32_f16(false, ap[0], false, b0, (short)0, oacc[t], false, false);
            oacc[t] = __builtin_amdgcn_wmma_f32_16x16x32_f16(false, ap[1], false, b1, (short)0, oacc[t], false, false);
        }

        if (more) scatterV(cur ^ 1);      // commit next V tile after the math
        wait_asynccnt0();                 // K async writes visible before barrier
        __syncthreads();
    }

    // ---- normalize + write merged-head output [B,S,D] fp32 ----
    const int b = bh >> 4, h = bh & 15;
    #pragma unroll
    for (int t = 0; t < 4; ++t) {
        #pragma unroll
        for (int i = 0; i < 8; ++i) {
            const int q = qbase + wv * 16 + hf * 8 + i;
            const int n = h * 64 + t * 16 + r;
            Op[((size_t)(b * SEQ + q)) * D_MODEL + n] = oacc[t][i] / lrow[i];
        }
    }
}

// ---------------------------------------------------------------------------
extern "C" void kernel_launch(void* const* d_in, const int* in_sizes, int n_in,
                              void* d_out, int out_size, void* d_ws, size_t ws_size,
                              hipStream_t stream)
{
    const float* q   = (const float*)d_in[0];
    const float* k   = (const float*)d_in[1];
    const float* v   = (const float*)d_in[2];
    const float* w_q = (const float*)d_in[3];
    const float* b_q = (const float*)d_in[4];
    const float* w_k = (const float*)d_in[5];
    const float* b_k = (const float*)d_in[6];
    const float* w_v = (const float*)d_in[7];
    const float* b_v = (const float*)d_in[8];
    const float* w_o = (const float*)d_in[9];
    const float* b_o = (const float*)d_in[10];

    char* ws = (char*)d_ws;
    _Float16* Qp = (_Float16*)(ws + 0);          // [B,H,S,64] fp16  (8 MB)
    _Float16* Kp = (_Float16*)(ws + 8388608);    // 8 MB
    _Float16* Vp = (_Float16*)(ws + 16777216);   // 8 MB
    float*    At = (float*)  (ws + 25165824);    // [B,S,D] fp32 attention out (16 MB)

    dim3 gg(D_MODEL / 64, MTOT / 128);   // (16, 32)
    dim3 bb(256);

    gemm_wmma_kernel<0><<<gg, bb, 0, stream>>>(q, w_q, b_q, (void*)Qp);
    gemm_wmma_kernel<0><<<gg, bb, 0, stream>>>(k, w_k, b_k, (void*)Kp);
    gemm_wmma_kernel<0><<<gg, bb, 0, stream>>>(v, w_v, b_v, (void*)Vp);

    attn_wmma_kernel<<<BATCH * NH * (SEQ / 64), 128, 0, stream>>>(Qp, Kp, Vp, At);

    gemm_wmma_kernel<1><<<gg, bb, 0, stream>>>(At, w_o, b_o, d_out);
}
